// TransformerRegressor_65644280152722
// MI455X (gfx1250) — compile-verified, matrix operand order checked by reading, not run on hardware
//
#include <hip/hip_runtime.h>
#include <hip/hip_bf16.h>

// ---------------- model constants ----------------
#define BATCH   2
#define SEQ     1024
#define DMODEL  256
#define NHEAD   8
#define HDIM    32
#define DFF     1024
#define NLAYERS 4
#define WINR    50          // radius
#define WINW    101         // 2*WINR+1
#define LN_EPS  1e-5f
#define TOKENS  (BATCH*SEQ) // 2048

// register blocking: each wave computes a (16*MT) x (16*NT) output tile
#define MT 2
#define NT 4

typedef __attribute__((ext_vector_type(16))) _Float16 v16h;
typedef __attribute__((ext_vector_type(8)))  float    v8f;

// ---------------- weight prep: f32 [K][N] -> f16 transposed [N][K] ----------------
__global__ void prep_wt(const float* __restrict__ w, _Float16* __restrict__ wt,
                        int K, int N) {
    int idx = blockIdx.x * blockDim.x + threadIdx.x;
    if (idx >= K * N) return;
    int k = idx / N;
    int n = idx % N;
    wt[(size_t)n * K + k] = (_Float16)w[(size_t)k * N + n];
}

// ---------------- input projection: [T,3] @ [3,256] + b  (writes f32 + f16 mirror) ----
__global__ void input_proj(const float* __restrict__ in, const float* __restrict__ W,
                           const float* __restrict__ b, float* __restrict__ x,
                           _Float16* __restrict__ x16) {
    int idx = blockIdx.x * blockDim.x + threadIdx.x;   // over TOKENS*DMODEL
    int t = idx / DMODEL;
    int n = idx % DMODEL;
    float acc = b[n];
    #pragma unroll
    for (int i = 0; i < 3; ++i)
        acc += in[t * 3 + i] * W[i * DMODEL + n];
    x[idx]   = acc;
    x16[idx] = (_Float16)acc;
}

// ---------------- WMMA GEMM: C[M,N] = A16[M,K] @ Wt[N,K] + bias ----------------
// one wave computes a 32x64 tile: MT x NT sub-tiles, 8 wmma per K-step of 32.
// C32 / C16 output ports are optional (pass nullptr to skip); uniform branches only.
__global__ void gemm_wmma(const _Float16* __restrict__ A, const _Float16* __restrict__ Wt,
                          const float* __restrict__ bias,
                          float* __restrict__ C32, _Float16* __restrict__ C16,
                          int M, int K, int N, int relu) {
    const int lane = threadIdx.x & 31;
    const int wave = (blockIdx.x * (blockDim.x >> 5)) + (threadIdx.x >> 5);
    const int tilesN = N / (16 * NT);
    const int tm = wave / tilesN;
    const int tn = wave % tilesN;

    const int l15   = lane & 15;
    const int khalf = (lane >> 4) * 8;            // 0 or 8

    const _Float16* arow[MT];
    const _Float16* brow[NT];
    int ncol[NT];
    #pragma unroll
    for (int i = 0; i < MT; ++i)
        arow[i] = A + (size_t)(tm * (16 * MT) + i * 16 + l15) * K;
    #pragma unroll
    for (int j = 0; j < NT; ++j) {
        ncol[j] = tn * (16 * NT) + j * 16 + l15;
        brow[j] = Wt + (size_t)ncol[j] * K;
    }

    v8f acc[MT][NT];
    #pragma unroll
    for (int i = 0; i < MT; ++i)
        #pragma unroll
        for (int j = 0; j < NT; ++j)
            acc[i][j] = (v8f){};

    for (int k0 = 0; k0 < K; k0 += 32) {
        if (k0 + 32 < K) {
            __builtin_prefetch(arow[0] + k0 + 32, 0, 3);   // global_prefetch_b8
            __builtin_prefetch(arow[MT - 1] + k0 + 32, 0, 3);
        }
        v16h a[MT], bm[NT];
        #pragma unroll
        for (int i = 0; i < MT; ++i) {
            #pragma unroll
            for (int e = 0; e < 8; ++e) a[i][e]     = arow[i][k0 + khalf + e];
            #pragma unroll
            for (int e = 0; e < 8; ++e) a[i][8 + e] = arow[i][k0 + 16 + khalf + e];
        }
        #pragma unroll
        for (int j = 0; j < NT; ++j) {
            #pragma unroll
            for (int e = 0; e < 8; ++e) bm[j][e]     = brow[j][k0 + khalf + e];
            #pragma unroll
            for (int e = 0; e < 8; ++e) bm[j][8 + e] = brow[j][k0 + 16 + khalf + e];
        }
        #pragma unroll
        for (int i = 0; i < MT; ++i)
            #pragma unroll
            for (int j = 0; j < NT; ++j)
                acc[i][j] = __builtin_amdgcn_wmma_f32_16x16x32_f16(
                    false, a[i], false, bm[j], (short)0, acc[i][j], false, false);
    }

    // epilogue: C/D layout — VGPR r holds row 8*(lane>>4)+r, column lane&15
    #pragma unroll
    for (int i = 0; i < MT; ++i) {
        const int rbase = tm * (16 * MT) + i * 16 + (lane >> 4) * 8;
        #pragma unroll
        for (int j = 0; j < NT; ++j) {
            const float bv = bias[ncol[j]];
            #pragma unroll
            for (int r = 0; r < 8; ++r) {
                float v = acc[i][j][r] + bv;
                if (relu) v = fmaxf(v, 0.0f);
                const size_t off = (size_t)(rbase + r) * N + ncol[j];
                if (C32) C32[off] = v;
                if (C16) C16[off] = (_Float16)v;
            }
        }
    }
}

// ---------------- banded attention: one wave per (b,h,s); f16 output ----------------
__global__ void banded_attn(const float* __restrict__ q, const float* __restrict__ k,
                            const float* __restrict__ v, const int* __restrict__ Lmask,
                            _Float16* __restrict__ out16) {
    __shared__ float probs[8 * 128];

    const int lane  = threadIdx.x & 31;
    const int winb  = threadIdx.x >> 5;
    const int wave  = blockIdx.x * 8 + winb;      // 0 .. B*NHEAD*SEQ-1
    const int b     = wave / (NHEAD * SEQ);
    const int rem   = wave % (NHEAD * SEQ);
    const int h     = rem / SEQ;
    const int s     = rem % SEQ;
    const int L     = Lmask[b];
    const float scale = 0.17677669529663687f;     // 1/sqrt(32)

    const float* qp = q + ((size_t)(b * SEQ + s)) * DMODEL + h * HDIM;
    float qreg[HDIM];
    #pragma unroll
    for (int d = 0; d < HDIM; ++d) qreg[d] = qp[d];

    float sc[4], ex[4];
    int   mk[4];
    float maxv = -3.0e38f;
    #pragma unroll
    for (int t = 0; t < 4; ++t) {
        const int j = lane + t * 32;
        sc[t] = -3.0e38f; mk[t] = 0;
        if (j < WINW) {
            const int idx = s - WINR + j;
            const int ic  = idx < 0 ? 0 : (idx > SEQ - 1 ? SEQ - 1 : idx);
            const int m   = (idx >= 0) && (idx < SEQ) && (s <= L) && (ic <= L);
            const float* kp = k + ((size_t)(b * SEQ + ic)) * DMODEL + h * HDIM;
            float dot = 0.0f;
            #pragma unroll
            for (int d = 0; d < HDIM; ++d) dot += qreg[d] * kp[d];
            sc[t] = m ? dot * scale : -1.0e9f;
            mk[t] = m;
            maxv  = fmaxf(maxv, sc[t]);
        }
    }
    #pragma unroll
    for (int o = 16; o > 0; o >>= 1) maxv = fmaxf(maxv, __shfl_xor(maxv, o));

    float ssum = 0.0f;
    #pragma unroll
    for (int t = 0; t < 4; ++t) {
        const int j = lane + t * 32;
        ex[t] = (j < WINW) ? __expf(sc[t] - maxv) : 0.0f;
        ssum += ex[t];
    }
    #pragma unroll
    for (int o = 16; o > 0; o >>= 1) ssum += __shfl_xor(ssum, o);
    const float inv = 1.0f / ssum;

    float* P = &probs[winb * 128];
    #pragma unroll
    for (int t = 0; t < 4; ++t) {
        const int j = lane + t * 32;
        P[j] = mk[t] ? ex[t] * inv : 0.0f;   // masked / out-of-window -> 0
    }
    __syncthreads();

    // lane = head channel
    float acc = 0.0f;
    for (int j = 0; j < WINW; ++j) {
        const int idx = s - WINR + j;
        const int ic  = idx < 0 ? 0 : (idx > SEQ - 1 ? SEQ - 1 : idx);
        acc += P[j] * v[((size_t)(b * SEQ + ic)) * DMODEL + h * HDIM + lane];
    }
    out16[((size_t)(b * SEQ + s)) * DMODEL + h * HDIM + lane] = (_Float16)acc;
}

// ---------------- fused residual + layernorm (writes f32 + f16 mirror) ----------------
// r indexed as r[b*rbstride + s*DMODEL + c]; rbstride==0 broadcasts batch 0 (a[0][None])
__global__ void residual_ln(const float* __restrict__ x, const float* __restrict__ r,
                            long rbstride, const float* __restrict__ g,
                            const float* __restrict__ bp, float* __restrict__ out,
                            _Float16* __restrict__ out16) {
    const int lane = threadIdx.x & 31;
    const int row  = blockIdx.x * (blockDim.x >> 5) + (threadIdx.x >> 5);
    const int b    = row / SEQ;
    const int s    = row % SEQ;

    const float* xr = x + (size_t)row * DMODEL;
    const float* rr = r + (size_t)b * rbstride + (size_t)s * DMODEL;

    float t[8];
    float sum = 0.0f;
    #pragma unroll
    for (int i = 0; i < 8; ++i) {
        t[i] = xr[lane * 8 + i] + rr[lane * 8 + i];
        sum += t[i];
    }
    #pragma unroll
    for (int o = 16; o > 0; o >>= 1) sum += __shfl_xor(sum, o);
    const float mu = sum * (1.0f / DMODEL);

    float vs = 0.0f;
    #pragma unroll
    for (int i = 0; i < 8; ++i) { const float d = t[i] - mu; vs += d * d; }
    #pragma unroll
    for (int o = 16; o > 0; o >>= 1) vs += __shfl_xor(vs, o);
    const float inv = rsqrtf(vs * (1.0f / DMODEL) + LN_EPS);

    #pragma unroll
    for (int i = 0; i < 8; ++i) {
        const int c = lane * 8 + i;
        const float v = (t[i] - mu) * inv * g[c] + bp[c];
        out[(size_t)row * DMODEL + c]   = v;
        out16[(size_t)row * DMODEL + c] = (_Float16)v;
    }
}

// ---------------- decode: [T,256] @ [256,20] + b ----------------
__global__ void decode(const float* __restrict__ x, const float* __restrict__ W,
                       const float* __restrict__ b, float* __restrict__ out) {
    const int idx = blockIdx.x * blockDim.x + threadIdx.x;  // TOKENS*20
    const int t = idx / 20;
    const int n = idx % 20;
    float acc = b[n];
    for (int kk = 0; kk < DMODEL; ++kk)
        acc += x[(size_t)t * DMODEL + kk] * W[kk * 20 + n];
    out[idx] = acc;
}

// ---------------- host ----------------
extern "C" void kernel_launch(void* const* d_in, const int* in_sizes, int n_in,
                              void* d_out, int out_size, void* d_ws, size_t ws_size,
                              hipStream_t stream) {
    (void)in_sizes; (void)n_in; (void)out_size; (void)ws_size;

    const float* in_coord = (const float*)d_in[0];
    const int*   pmask    = (const int*)  d_in[1];
    const float* inW      = (const float*)d_in[2];
    const float* inb      = (const float*)d_in[3];
    const float* decW     = (const float*)d_in[4 + 16 * NLAYERS + 0];
    const float* decb     = (const float*)d_in[4 + 16 * NLAYERS + 1];

    // bump allocator on workspace
    char* wp = (char*)d_ws;
    auto alloc = [&](size_t bytes) -> void* {
        void* r = wp;
        wp += (bytes + 255) & ~(size_t)255;
        return r;
    };
    float*    xA    = (float*)   alloc((size_t)TOKENS * DMODEL * 4);
    float*    xB    = (float*)   alloc((size_t)TOKENS * DMODEL * 4);
    _Float16* xA16  = (_Float16*)alloc((size_t)TOKENS * DMODEL * 2);
    _Float16* xB16  = (_Float16*)alloc((size_t)TOKENS * DMODEL * 2);
    float*    qb    = (float*)   alloc((size_t)TOKENS * DMODEL * 4);
    float*    kb    = (float*)   alloc((size_t)TOKENS * DMODEL * 4);
    float*    vb    = (float*)   alloc((size_t)TOKENS * DMODEL * 4);
    _Float16* att16 = (_Float16*)alloc((size_t)TOKENS * DMODEL * 2);
    float*    aproj = (float*)   alloc((size_t)TOKENS * DMODEL * 4);
    float*    ffb   = (float*)   alloc((size_t)TOKENS * DMODEL * 4);
    _Float16* h16   = (_Float16*)alloc((size_t)TOKENS * DFF * 2);

    _Float16* wqT[NLAYERS]; _Float16* wkT[NLAYERS]; _Float16* wvT[NLAYERS];
    _Float16* woT[NLAYERS]; _Float16* w1T[NLAYERS]; _Float16* w2T[NLAYERS];
    for (int l = 0; l < NLAYERS; ++l) {
        wqT[l] = (_Float16*)alloc((size_t)DMODEL * DMODEL * 2);
        wkT[l] = (_Float16*)alloc((size_t)DMODEL * DMODEL * 2);
        wvT[l] = (_Float16*)alloc((size_t)DMODEL * DMODEL * 2);
        woT[l] = (_Float16*)alloc((size_t)DMODEL * DMODEL * 2);
        w1T[l] = (_Float16*)alloc((size_t)DMODEL * DFF    * 2);
        w2T[l] = (_Float16*)alloc((size_t)DFF    * DMODEL * 2);
    }

    // weight prep (every call — deterministic)
    for (int l = 0; l < NLAYERS; ++l) {
        const int base = 4 + 16 * l;
        prep_wt<<<(DMODEL * DMODEL + 255) / 256, 256, 0, stream>>>(
            (const float*)d_in[base + 0], wqT[l], DMODEL, DMODEL);
        prep_wt<<<(DMODEL * DMODEL + 255) / 256, 256, 0, stream>>>(
            (const float*)d_in[base + 2], wkT[l], DMODEL, DMODEL);
        prep_wt<<<(DMODEL * DMODEL + 255) / 256, 256, 0, stream>>>(
            (const float*)d_in[base + 4], wvT[l], DMODEL, DMODEL);
        prep_wt<<<(DMODEL * DMODEL + 255) / 256, 256, 0, stream>>>(
            (const float*)d_in[base + 6], woT[l], DMODEL, DMODEL);
        prep_wt<<<(DMODEL * DFF    + 255) / 256, 256, 0, stream>>>(
            (const float*)d_in[base + 8], w1T[l], DMODEL, DFF);
        prep_wt<<<(DFF    * DMODEL + 255) / 256, 256, 0, stream>>>(
            (const float*)d_in[base + 10], w2T[l], DFF, DMODEL);
    }

    input_proj<<<(TOKENS * DMODEL) / 256, 256, 0, stream>>>(in_coord, inW, inb, xA, xA16);

    // gemm grids: waves = (M/(16*MT)) * (N/(16*NT)), 8 waves per 256-thread block
    const int gD = ((TOKENS / (16 * MT)) * (DMODEL / (16 * NT))) / 8; // 32 blocks
    const int gF = ((TOKENS / (16 * MT)) * (DFF    / (16 * NT))) / 8; // 128 blocks

    float*    x   = xA;   _Float16* x16  = xA16;
    float*    x2  = xB;   _Float16* x216 = xB16;
    for (int l = 0; l < NLAYERS; ++l) {
        const int base = 4 + 16 * l;
        const float* bq  = (const float*)d_in[base + 1];
        const float* bk  = (const float*)d_in[base + 3];
        const float* bv  = (const float*)d_in[base + 5];
        const float* bo  = (const float*)d_in[base + 7];
        const float* b1  = (const float*)d_in[base + 9];
        const float* b2  = (const float*)d_in[base + 11];
        const float* g1  = (const float*)d_in[base + 12];
        const float* be1 = (const float*)d_in[base + 13];
        const float* g2  = (const float*)d_in[base + 14];
        const float* be2 = (const float*)d_in[base + 15];

        gemm_wmma<<<gD, 256, 0, stream>>>(x16, wqT[l], bq, qb, (_Float16*)nullptr,
                                          TOKENS, DMODEL, DMODEL, 0);
        gemm_wmma<<<gD, 256, 0, stream>>>(x16, wkT[l], bk, kb, (_Float16*)nullptr,
                                          TOKENS, DMODEL, DMODEL, 0);
        gemm_wmma<<<gD, 256, 0, stream>>>(x16, wvT[l], bv, vb, (_Float16*)nullptr,
                                          TOKENS, DMODEL, DMODEL, 0);

        banded_attn<<<BATCH * NHEAD * SEQ / 8, 256, 0, stream>>>(qb, kb, vb, pmask, att16);

        gemm_wmma<<<gD, 256, 0, stream>>>(att16, woT[l], bo, aproj, (_Float16*)nullptr,
                                          TOKENS, DMODEL, DMODEL, 0);

        // residual uses a[0] broadcast over batch (rbstride = 0)
        residual_ln<<<TOKENS / 8, 256, 0, stream>>>(x, aproj, 0L, g1, be1, x2, x216);

        gemm_wmma<<<gF, 256, 0, stream>>>(x216, w1T[l], b1, (float*)nullptr, h16,
                                          TOKENS, DMODEL, DFF, 1);
        gemm_wmma<<<gD, 256, 0, stream>>>(h16, w2T[l], b2, ffb, (_Float16*)nullptr,
                                          TOKENS, DFF, DMODEL, 0);

        residual_ln<<<TOKENS / 8, 256, 0, stream>>>(x2, ffb, (long)SEQ * DMODEL, g2, be2,
                                                    x, x16);
    }

    decode<<<(TOKENS * 20) / 256, 256, 0, stream>>>(x, decW, decb, (float*)d_out);
}
